// MultiheadSelfAttention_40699110097313
// MI455X (gfx1250) — matrix operand order repaired and checked by reading, not compiled, not run on hardware
//
#include <hip/hip_runtime.h>
#include <hip/hip_bf16.h>

// ---------------- problem constants ----------------
#define B_   4
#define S_   2048
#define D_   1024
#define H_   16
#define HD_  64
#define SCALE_ 0.03125f            // D^-0.5 = 1/32  (reference scales by embed dim)
#define MROWS_ (B_*S_)             // 8192
#define QSZ_  ((size_t)B_*H_*S_*HD_)  // 8,388,608 elements per Q/K/V tensor

typedef __attribute__((ext_vector_type(16))) __bf16 v16bf;
typedef __attribute__((ext_vector_type(8)))  __bf16 v8bf;
typedef __attribute__((ext_vector_type(4)))  __bf16 v4bf;
typedef __attribute__((ext_vector_type(8)))  float  v8f;
typedef __attribute__((ext_vector_type(4)))  float  v4f;

union FragBF { v16bf v; v8bf h[2]; };

static __device__ inline v8f zero8() {
  v8f z = {0.f,0.f,0.f,0.f,0.f,0.f,0.f,0.f};
  return z;
}

static __device__ inline v8f wmma_bf16(const FragBF& a, const FragBF& b, v8f c) {
  // emits v_wmma_f32_16x16x32_bf16
  return __builtin_amdgcn_wmma_f32_16x16x32_bf16(
      /*neg_a=*/false, a.v, /*neg_b=*/false, b.v,
      /*c_mod=*/(short)0, c, /*reuse_a=*/false, /*reuse_b=*/false);
}

// A-fragment (16x32 bf16, M x K): lane = half*16+l16 holds row M=l16,
// K chunks at 8*half and 8*half+16 (each 8 contiguous bf16 = 16B).
static __device__ inline FragBF load_fragA(const __bf16* base, int row_stride,
                                           int m0, int k, int l16, int half) {
  FragBF a;
  const __bf16* p = base + (size_t)(m0 + l16) * row_stride + k + half * 8;
  a.h[0] = *(const v8bf*)(p);
  a.h[1] = *(const v8bf*)(p + 16);
  return a;
}

// B-fragment (32x16 bf16, K x N) built from a row-major W so B = W^T tile:
// lane holds column n=l16 -> W row (n0+l16), K chunks at 8*half and 8*half+16.
static __device__ inline FragBF load_fragBT(const __bf16* base, int row_stride,
                                            int n0, int k, int l16, int half) {
  FragBF b;
  const __bf16* p = base + (size_t)(n0 + l16) * row_stride + k + half * 8;
  b.h[0] = *(const v8bf*)(p);
  b.h[1] = *(const v8bf*)(p + 16);
  return b;
}

// ---------------- fp32 -> bf16 conversion (vectorized x4) ----------------
__global__ void k_cvt_bf16(const float* __restrict__ src,
                           __bf16* __restrict__ dst, int n4) {
  int i = blockIdx.x * blockDim.x + threadIdx.x;
  if (i < n4) {
    v4f f = ((const v4f*)src)[i];
    v4bf o;
    o.x = (__bf16)f.x; o.y = (__bf16)f.y; o.z = (__bf16)f.z; o.w = (__bf16)f.w;
    ((v4bf*)dst)[i] = o;
  }
}

// ---------------- QKV projection GEMM ----------------
// C[8192 x 3072] = Xb[8192 x 1024] @ Wqkv^T + bqkv ; scattered to head layout.
// Q,K stored [b][h][s][64] ; V stored transposed [b][h][64][s].
// Wave tile: 32(M) x 64(N): 8 accum frags, per k-step 2 A + 4 B loads -> 8 WMMAs.
__global__ void k_qkv_gemm(const __bf16* __restrict__ xb,
                           const __bf16* __restrict__ wb,   // [3072][1024] row-major
                           const float*  __restrict__ bias, // [3072]
                           __bf16* __restrict__ qkvb) {
  const int lane = threadIdx.x & 31;
  const int wid  = threadIdx.x >> 5;
  const int task = blockIdx.x * 8 + wid;          // 12288 tasks = 256 Mtiles(32) * 48 Nstrips
  const int tm   = task / 48;
  const int ns   = task % 48;
  const int m0   = tm * 32, n0 = ns * 64;
  const int half = lane >> 4, l16 = lane & 15;

  v8f acc[2][4] = { { zero8(), zero8(), zero8(), zero8() },
                    { zero8(), zero8(), zero8(), zero8() } };

  for (int k = 0; k < D_; k += 32) {
    // issue all fragment loads for this k-step first (staged load waits)
    FragBF a0 = load_fragA(xb, D_, m0,      k, l16, half);
    FragBF a1 = load_fragA(xb, D_, m0 + 16, k, l16, half);
    FragBF b0 = load_fragBT(wb, D_, n0,      k, l16, half);
    FragBF b1 = load_fragBT(wb, D_, n0 + 16, k, l16, half);
    FragBF b2 = load_fragBT(wb, D_, n0 + 32, k, l16, half);
    FragBF b3 = load_fragBT(wb, D_, n0 + 48, k, l16, half);
    acc[0][0] = wmma_bf16(a0, b0, acc[0][0]);
    acc[1][0] = wmma_bf16(a1, b0, acc[1][0]);
    acc[0][1] = wmma_bf16(a0, b1, acc[0][1]);
    acc[1][1] = wmma_bf16(a1, b1, acc[1][1]);
    acc[0][2] = wmma_bf16(a0, b2, acc[0][2]);
    acc[1][2] = wmma_bf16(a1, b2, acc[1][2]);
    acc[0][3] = wmma_bf16(a0, b3, acc[0][3]);
    acc[1][3] = wmma_bf16(a1, b3, acc[1][3]);
  }

#pragma unroll
  for (int r = 0; r < 2; ++r) {
#pragma unroll
    for (int t = 0; t < 4; ++t) {
      const int n     = n0 + t * 16 + l16;
      const int which = n >> 10;          // 0=q,1=k,2=v
      const int d     = n & 1023;
      const int h     = d >> 6, hd = d & 63;
      const float bv  = bias[n];
#pragma unroll
      for (int i = 0; i < 8; ++i) {
        const int m  = m0 + r * 16 + i + half * 8;
        const int bb = m >> 11;           // / 2048
        const int s  = m & 2047;
        const float val = acc[r][t][i] + bv;
        size_t off;
        if (which == 2)
          off = 2 * QSZ_ + (((size_t)(bb * H_ + h) * HD_ + hd) * S_ + s);
        else
          off = (size_t)which * QSZ_ + (((size_t)(bb * H_ + h) * S_ + s) * HD_ + hd);
        qkvb[off] = (__bf16)val;
      }
    }
  }
}

// ---------------- causal flash attention (one wave per 16-query tile) ------
__global__ void k_attn(const __bf16* __restrict__ qkvb,
                       __bf16* __restrict__ attnb) {
  __shared__ float  Ss[16][32];
  __shared__ __bf16 Ps[16][32];
  __shared__ float  alpha_s[16];
  __shared__ float  mrow[16];
  __shared__ float  lrow[16];

  const int lane = threadIdx.x;        // blockDim == 32, one wave
  const int half = lane >> 4, l16 = lane & 15;
  int idx = blockIdx.x;
  const int qb = idx & 127;  idx >>= 7;   // S/16 = 128 query tiles
  const int h  = idx & 15;   idx >>= 4;
  const int b  = idx;

  const __bf16* Q  = qkvb              + ((size_t)(b * H_ + h) * S_) * HD_;
  const __bf16* Kp = qkvb + QSZ_       + ((size_t)(b * H_ + h) * S_) * HD_;
  const __bf16* Vt = qkvb + 2 * QSZ_   + ((size_t)(b * H_ + h) * HD_) * S_;

  // Q fragments: K-dim = HD = 64 -> 2 WMMA k-steps
  FragBF qa[2];
#pragma unroll
  for (int s = 0; s < 2; ++s)
    qa[s] = load_fragA(Q, HD_, qb * 16, s * 32, l16, half);

  v8f o[4] = { zero8(), zero8(), zero8(), zero8() };
  if (lane < 16) { mrow[lane] = -1e30f; lrow[lane] = 0.f; }
  __syncthreads();

  const int q16 = qb * 16;
  const int nkb = (q16 + 15) / 32 + 1;     // 32-key blocks covering keys <= q16+15

  for (int kb = 0; kb < nkb; ++kb) {
    const int kbase = kb * 32;

    // ---- scores: S[16q x 32k] = Q @ K^T
    v8f sc[2] = { zero8(), zero8() };
#pragma unroll
    for (int t = 0; t < 2; ++t) {
#pragma unroll
      for (int s = 0; s < 2; ++s) {
        FragBF bfrag = load_fragBT(Kp, HD_, kbase + t * 16, s * 32, l16, half);
        sc[t] = wmma_bf16(qa[s], bfrag, sc[t]);
      }
    }
#pragma unroll
    for (int t = 0; t < 2; ++t)
#pragma unroll
      for (int i = 0; i < 8; ++i)
        Ss[i + half * 8][t * 16 + l16] = sc[t][i];
    __syncthreads();

    // ---- online causal softmax, lanes 0..15 own one row each
    if (lane < 16) {
      const int row = lane;
      const int qi  = q16 + row;
      const float mold = mrow[row];
      float vbuf[32];
      float bmax = -1e30f;
#pragma unroll
      for (int j = 0; j < 32; ++j) {
        const bool valid = (kbase + j) <= qi;
        const float v = valid ? Ss[row][j] * SCALE_ : -1e30f;
        vbuf[j] = v;
        bmax = fmaxf(bmax, v);
      }
      const float mnew = fmaxf(mold, bmax);
      const float al   = __expf(mold - mnew);
      float sum = 0.f;
#pragma unroll
      for (int j = 0; j < 32; ++j) {
        const float p = (vbuf[j] > -1e29f) ? __expf(vbuf[j] - mnew) : 0.f;
        sum += p;
        Ps[row][j] = (__bf16)p;
      }
      mrow[row]    = mnew;
      lrow[row]    = lrow[row] * al + sum;
      alpha_s[row] = al;
    }
    __syncthreads();

    // ---- rescale accumulators by alpha(row)
    float al8[8];
#pragma unroll
    for (int i = 0; i < 8; ++i) al8[i] = alpha_s[i + half * 8];
#pragma unroll
    for (int g = 0; g < 4; ++g)
#pragma unroll
      for (int i = 0; i < 8; ++i) o[g][i] *= al8[i];

    // ---- P fragment (A-layout) from LDS
    FragBF pf;
    const __bf16* pp = &Ps[l16][half * 8];
    pf.h[0] = *(const v8bf*)(pp);
    pf.h[1] = *(const v8bf*)(pp + 16);

    // ---- O += P @ V  (V pre-transposed: contiguous B-frag loads)
#pragma unroll
    for (int g = 0; g < 4; ++g) {
      FragBF vf = load_fragBT(Vt, S_, 0, 0, l16, half);
      // recompute pointer properly: rows are hd (g*16+l16), cols are keys
      const __bf16* vp = Vt + (size_t)(g * 16 + l16) * S_ + kbase + half * 8;
      vf.h[0] = *(const v8bf*)(vp);
      vf.h[1] = *(const v8bf*)(vp + 16);
      o[g] = wmma_bf16(pf, vf, o[g]);
    }
    __syncthreads();
  }

  // ---- finalize: O / l, store to [b][s][h*64+hd] (bf16)
  float l8[8];
#pragma unroll
  for (int i = 0; i < 8; ++i) l8[i] = lrow[i + half * 8];
#pragma unroll
  for (int g = 0; g < 4; ++g) {
#pragma unroll
    for (int i = 0; i < 8; ++i) {
      const int s    = q16 + i + half * 8;
      const int dcol = h * HD_ + g * 16 + l16;
      attnb[((size_t)b * S_ + s) * D_ + dcol] = (__bf16)(o[g][i] / l8[i]);
    }
  }
}

// ---------------- lift GEMM: out[8192 x 1024] = Attn @ Wlift^T + blift ------
// Wave tile: 32(M) x 64(N), same structure as k_qkv_gemm.
__global__ void k_lift_gemm(const __bf16* __restrict__ ab,
                            const __bf16* __restrict__ wb,   // [1024][1024]
                            const float*  __restrict__ bias, // [1024]
                            float* __restrict__ out) {
  const int lane = threadIdx.x & 31;
  const int wid  = threadIdx.x >> 5;
  const int task = blockIdx.x * 8 + wid;          // 4096 tasks = 256 Mtiles(32) * 16 Nstrips
  const int tm   = task / 16;
  const int ns   = task % 16;
  const int m0   = tm * 32, n0 = ns * 64;
  const int half = lane >> 4, l16 = lane & 15;

  v8f acc[2][4] = { { zero8(), zero8(), zero8(), zero8() },
                    { zero8(), zero8(), zero8(), zero8() } };

  for (int k = 0; k < D_; k += 32) {
    FragBF a0 = load_fragA(ab, D_, m0,      k, l16, half);
    FragBF a1 = load_fragA(ab, D_, m0 + 16, k, l16, half);
    FragBF b0 = load_fragBT(wb, D_, n0,      k, l16, half);
    FragBF b1 = load_fragBT(wb, D_, n0 + 16, k, l16, half);
    FragBF b2 = load_fragBT(wb, D_, n0 + 32, k, l16, half);
    FragBF b3 = load_fragBT(wb, D_, n0 + 48, k, l16, half);
    acc[0][0] = wmma_bf16(a0, b0, acc[0][0]);
    acc[1][0] = wmma_bf16(a1, b0, acc[1][0]);
    acc[0][1] = wmma_bf16(a0, b1, acc[0][1]);
    acc[1][1] = wmma_bf16(a1, b1, acc[1][1]);
    acc[0][2] = wmma_bf16(a0, b2, acc[0][2]);
    acc[1][2] = wmma_bf16(a1, b2, acc[1][2]);
    acc[0][3] = wmma_bf16(a0, b3, acc[0][3]);
    acc[1][3] = wmma_bf16(a1, b3, acc[1][3]);
  }

#pragma unroll
  for (int r = 0; r < 2; ++r) {
#pragma unroll
    for (int t = 0; t < 4; ++t) {
      const int n    = n0 + t * 16 + l16;
      const float bv = bias[n];
#pragma unroll
      for (int i = 0; i < 8; ++i) {
        const int m = m0 + r * 16 + i + half * 8;
        out[(size_t)m * D_ + n] = acc[r][t][i] + bv;
      }
    }
  }
}

// ---------------- host-side launcher ----------------
extern "C" void kernel_launch(void* const* d_in, const int* in_sizes, int n_in,
                              void* d_out, int out_size, void* d_ws, size_t ws_size,
                              hipStream_t stream) {
  const float* x     = (const float*)d_in[0];   // [4,2048,1024]
  const float* Wqkv  = (const float*)d_in[1];   // [3072,1024]
  const float* bqkv  = (const float*)d_in[2];   // [3072]
  const float* Wlift = (const float*)d_in[3];   // [1024,1024]
  const float* blift = (const float*)d_in[4];   // [1024]
  float* out = (float*)d_out;

  // workspace carve-up (bytes)
  char* ws = (char*)d_ws;
  const size_t XN  = (size_t)MROWS_ * D_;       // 8,388,608
  const size_t WQN = (size_t)3 * D_ * D_;       // 3,145,728
  const size_t WLN = (size_t)D_ * D_;           // 1,048,576
  __bf16* xb    = (__bf16*)(ws);                                     // 16 MB
  __bf16* wqb   = (__bf16*)(ws + XN * 2);                            //  6 MB
  __bf16* wlb   = (__bf16*)(ws + XN * 2 + WQN * 2);                  //  2 MB
  __bf16* qkvb  = (__bf16*)(ws + XN * 2 + WQN * 2 + WLN * 2);        // 48 MB
  __bf16* attnb = (__bf16*)(ws + XN * 2 + WQN * 2 + WLN * 2 + 3 * QSZ_ * 2); // 16 MB

  // 1) fp32 -> bf16 conversions
  k_cvt_bf16<<<(int)(XN / 4 / 256),  256, 0, stream>>>(x,     xb,  (int)(XN / 4));
  k_cvt_bf16<<<(int)(WQN / 4 / 256), 256, 0, stream>>>(Wqkv,  wqb, (int)(WQN / 4));
  k_cvt_bf16<<<(int)(WLN / 4 / 256), 256, 0, stream>>>(Wlift, wlb, (int)(WLN / 4));

  // 2) QKV projection GEMM (WMMA): 12288 wave-tiles / 8 waves per block
  k_qkv_gemm<<<1536, 256, 0, stream>>>(xb, wqb, bqkv, qkvb);

  // 3) causal flash attention: one wave per (b, h, 16-query tile)
  k_attn<<<B_ * H_ * (S_ / 16), 32, 0, stream>>>(qkvb, attnb);

  // 4) lift GEMM (WMMA), fp32 output
  k_lift_gemm<<<512, 256, 0, stream>>>(attnb, wlb, blift, out);
}
